// QLSTM_65481071397983
// MI455X (gfx1250) — matrix operand order repaired and checked by reading, not compiled, not run on hardware
//
#include <hip/hip_runtime.h>
#include <hip/hip_bf16.h>

// ---- problem constants (match reference) ----
#define T_STEPS 512
#define BATCH   64
#define IN      512
#define HID     1024
#define KTOT    (IN + HID)   // 1536
#define GATES   (4 * HID)    // 4096

typedef __bf16 bf16_t;
typedef __attribute__((ext_vector_type(16))) __bf16 v16bf;
typedef __attribute__((ext_vector_type(8)))  __bf16 v8bf;
typedef __attribute__((ext_vector_type(8)))  float  v8f;

// ---------------- fp32 -> bf16 convert (W, X) ----------------
__global__ void cvt_f32_bf16_kernel(const float* __restrict__ src,
                                    bf16_t* __restrict__ dst, int n) {
    int i = blockIdx.x * blockDim.x + threadIdx.x;
    int stride = gridDim.x * blockDim.x;
    for (; i < n; i += stride) dst[i] = (bf16_t)src[i];
}

// ---------------- zero hx (bf16) and cx (f32) ----------------
__global__ void init_state_kernel(bf16_t* __restrict__ hx0,
                                  float* __restrict__ cx, int n) {
    int i = blockIdx.x * blockDim.x + threadIdx.x;
    int stride = gridDim.x * blockDim.x;
    for (; i < n; i += stride) { hx0[i] = (bf16_t)0.0f; cx[i] = 0.0f; }
}

// ---------------- one LSTM timestep, fused gates+cell ----------------
// grid = (BATCH/16, HID/16), block = 128 threads = 4 waves (wave32).
// Wave g (0..3) computes gate g's 16x16 f32 tile with v_wmma_f32_16x16x32_bf16,
// K = 1536 (x part then h part). All four gates for the same (b,h) live in one
// workgroup, so the sigmoid/tanh cell update is fused via LDS.
__global__ __launch_bounds__(128) void lstm_step_kernel(
    int t,
    const bf16_t* __restrict__ Xbf,    // [T, B, IN]  bf16
    const bf16_t* __restrict__ Wbf,    // [GATES, KTOT] bf16 (row-major = B-matrix order)
    const float*  __restrict__ bias,   // [GATES] f32
    const bf16_t* __restrict__ hx_in,  // [B, HID] bf16 (read)
    bf16_t*       __restrict__ hx_out, // [B, HID] bf16 (write)
    float*        __restrict__ cx,     // [B, HID] f32 (read+write)
    float*        __restrict__ out)    // [T, B, HID] f32 stacked output
{
    __shared__ float g4[4][16][16];    // [gate][m][h-col] pre-activations

    const int tid    = threadIdx.x;
    const int lane   = tid & 31;
    const int gi     = tid >> 5;        // wave id == gate id (0..3)
    const int laneLo = lane & 15;
    const int laneHi = lane >> 4;       // 0 or 1

    const int rt = blockIdx.x;          // batch tile: rows rt*16 .. +15
    const int hg = blockIdx.y;          // h tile:     cols hg*16 .. +15

    const int row  = rt * 16 + laneLo;              // A-matrix M (lanes repeat M twice)
    const int ncol = gi * HID + hg * 16 + laneLo;   // B/D-matrix N (global gate column)

    const bf16_t* __restrict__ xrow = Xbf + ((size_t)t * BATCH + row) * IN;
    const bf16_t* __restrict__ hrow = hx_in + (size_t)row * HID;
    const bf16_t* __restrict__ wrow = Wbf + (size_t)ncol * KTOT;

    v8f acc = {};

    // ---- K over input part: k in [0, 512) ----
    for (int k0 = 0; k0 < IN; k0 += 32) {
        union { v16bf v; v8bf h[2]; } a;
        // A k order used here (contiguous): lanes 0-15 hold K k0+[0..15],
        //                  lanes 16-31 hold K k0+16+[0..15]
        a.h[0] = *(const v8bf*)(xrow + k0 +     laneHi * 16);
        a.h[1] = *(const v8bf*)(xrow + k0 + 8 + laneHi * 16);
        // B k order used here (same as A): lane n holds col n, 16 contiguous K (hi lanes +16)
        v16bf bmat = *(const v16bf*)(wrow + k0 + laneHi * 16);
        acc = __builtin_amdgcn_wmma_f32_16x16x32_bf16(
            false, a.v, false, bmat, (short)0, acc, false, false);
    }

    // ---- K over hidden part: k in [512, 1536) ----
    for (int k0 = 0; k0 < HID; k0 += 32) {
        __builtin_prefetch(wrow + IN + k0 + 512, 0, 1);   // keep weight stream hot
        union { v16bf v; v8bf h[2]; } a;
        a.h[0] = *(const v8bf*)(hrow + k0 +     laneHi * 16);
        a.h[1] = *(const v8bf*)(hrow + k0 + 8 + laneHi * 16);
        v16bf bmat = *(const v16bf*)(wrow + IN + k0 + laneHi * 16);
        acc = __builtin_amdgcn_wmma_f32_16x16x32_bf16(
            false, a.v, false, bmat, (short)0, acc, false, false);
    }

    // ---- stage this gate's 16x16 f32 tile to LDS (C/D layout, ISA 7.12.2) ----
#pragma unroll
    for (int r = 0; r < 8; ++r) {
        const int m = laneHi ? (r + 8) : r;   // lanes 16-31 carry M = r+8
        g4[gi][m][laneLo] = acc[r];
    }
    __syncthreads();

    // ---- fused LSTM cell update: 256 elements, 128 threads ----
    for (int idx = tid; idx < 16 * 16; idx += 128) {
        const int m = idx >> 4;
        const int c = idx & 15;
        const int b = rt * 16 + m;
        const int h = hg * 16 + c;

        const float fv = g4[0][m][c] + bias[h];
        const float iv = g4[1][m][c] + bias[HID + h];
        const float gv = g4[2][m][c] + bias[2 * HID + h];
        const float ov = g4[3][m][c] + bias[3 * HID + h];

        const float fs = 1.0f / (1.0f + __expf(-fv));
        const float is = 1.0f / (1.0f + __expf(-iv));
        const float gt = tanhf(gv);
        const float os = 1.0f / (1.0f + __expf(-ov));

        const size_t sidx = (size_t)b * HID + h;
        const float cn = fs * cx[sidx] + is * gt;
        const float hn = os * tanhf(cn);

        cx[sidx]     = cn;
        hx_out[sidx] = (bf16_t)hn;                         // next step's A-matrix
        out[((size_t)t * BATCH + b) * HID + h] = hn;       // stacked output (f32)
    }
}

// ---------------- write (hx, cx) tail of the output tuple ----------------
__global__ void finalize_kernel(const float* __restrict__ stacked,
                                const float* __restrict__ cx,
                                float* __restrict__ out_hx,
                                float* __restrict__ out_cx, int n) {
    int i = blockIdx.x * blockDim.x + threadIdx.x;
    int stride = gridDim.x * blockDim.x;
    for (; i < n; i += stride) {
        out_hx[i] = stacked[(size_t)(T_STEPS - 1) * BATCH * HID + i];
        out_cx[i] = cx[i];
    }
}

extern "C" void kernel_launch(void* const* d_in, const int* in_sizes, int n_in,
                              void* d_out, int out_size, void* d_ws, size_t ws_size,
                              hipStream_t stream) {
    (void)in_sizes; (void)n_in; (void)out_size;

    const float* X = (const float*)d_in[0];   // [T, B, IN]
    const float* W = (const float*)d_in[1];   // [GATES, KTOT]
    const float* b = (const float*)d_in[2];   // [GATES]
    float* out = (float*)d_out;               // stacked ++ hx ++ cx

    // ---- workspace carve-out (~47 MB total) ----
    char* ws = (char*)d_ws;
    size_t off = 0;
    auto carve = [&](size_t bytes) -> void* {
        off = (off + 255) & ~(size_t)255;
        void* p = ws + off;
        off += bytes;
        return p;
    };
    bf16_t* Wbf = (bf16_t*)carve((size_t)GATES * KTOT * sizeof(bf16_t));       // 12 MB
    bf16_t* Xbf = (bf16_t*)carve((size_t)T_STEPS * BATCH * IN * sizeof(bf16_t)); // 32 MB
    bf16_t* hxA = (bf16_t*)carve((size_t)BATCH * HID * sizeof(bf16_t));
    bf16_t* hxB = (bf16_t*)carve((size_t)BATCH * HID * sizeof(bf16_t));
    float*  cxv = (float*) carve((size_t)BATCH * HID * sizeof(float));
    (void)ws_size;

    // one-time (per launch) conversions + state init — deterministic every call
    cvt_f32_bf16_kernel<<<2048, 256, 0, stream>>>(W, Wbf, GATES * KTOT);
    cvt_f32_bf16_kernel<<<4096, 256, 0, stream>>>(X, Xbf, T_STEPS * BATCH * IN);
    init_state_kernel<<<64, 256, 0, stream>>>(hxA, cxv, BATCH * HID);

    // serial recurrence: 512 dependent step launches (graph-captured by harness)
    dim3 grid(BATCH / 16, HID / 16);  // (4, 64) = 256 workgroups/step
    for (int t = 0; t < T_STEPS; ++t) {
        const bf16_t* hin  = (t & 1) ? hxB : hxA;  // double-buffer hx: avoids
        bf16_t*       hout = (t & 1) ? hxA : hxB;  // cross-workgroup RAW race
        lstm_step_kernel<<<grid, 128, 0, stream>>>(t, Xbf, Wbf, b, hin, hout, cxv, out);
    }

    float* out_hx = out + (size_t)T_STEPS * BATCH * HID;
    float* out_cx = out_hx + (size_t)BATCH * HID;
    finalize_kernel<<<256, 256, 0, stream>>>(out, cxv, out_hx, out_cx, BATCH * HID);
}
